// UpTransformer_cross_46377056862436
// MI455X (gfx1250) — matrix operand order repaired and checked by reading, not compiled, not run on hardware
//
#include <hip/hip_runtime.h>
#include <hip/hip_bf16.h>
#include <math.h>

// ---------------------------------------------------------------------------
// MI455X / gfx1250 implementation of the UpTransformer cross-attention block.
// GEMMs on v_wmma_f32_16x16x32_bf16; tile staging via CDNA5 async-to-LDS DMA
// (global_load_async_to_lds_b128 + s_wait_asynccnt) with double buffering.
// ---------------------------------------------------------------------------

typedef __attribute__((ext_vector_type(16))) __bf16 v16bf;
typedef __attribute__((ext_vector_type(8)))  float  v8f;

#define B_  256
#define N_  256
#define C_  256
#define H_  4
#define DH_ 64
#define FF_ 1024
#define T_  (B_ * N_)          // 65536 token rows

// ----------------------------- small helpers -------------------------------

static __device__ __forceinline__ __bf16 f2bf(float f) {
  unsigned u = __builtin_bit_cast(unsigned, f);
  unsigned r = (u + 0x7FFFu + ((u >> 16) & 1u)) >> 16;  // round-to-nearest-even
  unsigned short s = (unsigned short)r;
  return __builtin_bit_cast(__bf16, s);
}

// CDNA5 async DMA: copy 16 bytes from global to LDS, tracked by ASYNCcnt.
static __device__ __forceinline__ void async_b128(const void* g, const void* l) {
  unsigned loff = (unsigned)(unsigned long long)
      (__attribute__((address_space(3))) const void*)l;   // LDS byte offset
  unsigned long long ga = (unsigned long long)g;
  asm volatile("global_load_async_to_lds_b128 %0, %1, off"
               :: "v"(loff), "v"(ga) : "memory");
}
static __device__ __forceinline__ void wait_async0() {
  asm volatile("s_wait_asynccnt 0x0" ::: "memory");
}

union VB { v16bf v; unsigned u[8]; unsigned short s[16]; };

// A fragment: 16x32 bf16 (rows row0.., K = kOff..kOff+31). Contiguous k = 16*(lane/16) + s.
static __device__ __forceinline__ v16bf load_a(const __bf16* lds, int row0, int kOff, int ld) {
  const unsigned* lu = (const unsigned*)lds;
  const int lane  = threadIdx.x & 31;
  const int m     = row0 + (lane & 15);
  const int half8 = (lane >> 4) << 4;
  const int ldDw  = ld >> 1;
  VB r;
#pragma unroll
  for (int v = 0; v < 8; ++v) {
    int k0 = half8 + (v << 1);   // even
    r.u[v] = lu[m * ldDw + ((kOff + k0) >> 1)];
  }
  return r.v;
}

// B fragment from W^T: LDS holds W rows (= output cols) x K, B[k][n] = W[n][k].
static __device__ __forceinline__ v16bf load_b_wt(const __bf16* lds, int col0, int kOff, int ld) {
  const unsigned* lu = (const unsigned*)lds;
  const int lane = threadIdx.x & 31;
  const int n    = col0 + (lane & 15);
  const int kb   = (lane >> 4) << 4;
  const int ldDw = ld >> 1;
  VB r;
#pragma unroll
  for (int v = 0; v < 8; ++v)
    r.u[v] = lu[n * ldDw + ((kOff + kb + (v << 1)) >> 1)];
  return r.v;
}

// B fragment from a row-major KxN tile in LDS (B[k][n] at lds[k*ld+n]).
static __device__ __forceinline__ v16bf load_b_rm(const __bf16* lds, int kOff, int col0, int ld) {
  const unsigned short* ls = (const unsigned short*)lds;
  const int lane = threadIdx.x & 31;
  const int n    = col0 + (lane & 15);
  const int kb   = kOff + ((lane >> 4) << 4);
  VB r;
#pragma unroll
  for (int v = 0; v < 8; ++v) {
    int k = kb + (v << 1);
    unsigned lo = ls[k * ld + n];
    unsigned hi = ls[(k + 1) * ld + n];
    r.u[v] = lo | (hi << 16);
  }
  return r.v;
}

static __device__ __forceinline__ v8f wmma_bf16(v16bf a, v16bf b, v8f c) {
  return __builtin_amdgcn_wmma_f32_16x16x32_bf16(false, a, false, b, (short)0, c, false, false);
}

static __device__ __forceinline__ float half16_max(float v) {
#pragma unroll
  for (int off = 8; off; off >>= 1) v = fmaxf(v, __shfl_xor(v, off, 32));
  return v;
}
static __device__ __forceinline__ float half16_sum(float v) {
#pragma unroll
  for (int off = 8; off; off >>= 1) v += __shfl_xor(v, off, 32);
  return v;
}

// Stage one 128x32 bf16 tile (row-major, ld K) into padded LDS (ld 40) via DMA.
static __device__ __forceinline__ void stage_tile(const __bf16* __restrict__ G,
                                                  int rowB, int K, int k0,
                                                  __bf16* dst, int tid) {
#pragma unroll
  for (int i = tid; i < 512; i += 256) {       // 512 x 16B chunks
    int r = i >> 2, sg = i & 3;
    async_b128(G + (size_t)(rowB + r) * K + k0 + sg * 8, dst + r * 40 + sg * 8);
  }
}

// ----------------------------- tiled WMMA GEMM -----------------------------
// C(M,N) = act(A(M,K) @ W(N,K)^T + bias). 128x128 block, 8 waves, each wave
// 32x64 (2x4 WMMA tiles). Double-buffered async-to-LDS staging.

template<int ACT, bool WF32, bool WBF16>
__global__ __launch_bounds__(256)
void k_gemm(const __bf16* __restrict__ A, const __bf16* __restrict__ W,
            const float* __restrict__ bias,
            float* __restrict__ Cf, __bf16* __restrict__ Cb,
            int M, int Nn, int K)
{
  __shared__ __bf16 sA[2][128 * 40];
  __shared__ __bf16 sB[2][128 * 40];
  const int tid     = threadIdx.x;
  const int rowBase = blockIdx.y * 128;
  const int colBase = blockIdx.x * 128;
  const int wv = tid >> 5;
  const int wm = wv & 3;            // 4 waves across M
  const int wn = wv >> 2;           // 2 waves across N
  const int lane = tid & 31;

  v8f acc[2][4];
#pragma unroll
  for (int i = 0; i < 2; ++i)
#pragma unroll
    for (int j = 0; j < 4; ++j) acc[i][j] = v8f{};

  // prologue: stage first K-tile
  stage_tile(A, rowBase, K, 0, sA[0], tid);
  stage_tile(W, colBase, K, 0, sB[0], tid);

  int p = 0;
  for (int k0 = 0; k0 < K; k0 += 32) {
    wait_async0();
    __syncthreads();
    if (k0 + 32 < K) {               // overlap next tile's DMA with WMMA
      stage_tile(A, rowBase, K, k0 + 32, sA[p ^ 1], tid);
      stage_tile(W, colBase, K, k0 + 32, sB[p ^ 1], tid);
    }
    v16bf a0 = load_a(sA[p], wm * 32,      0, 40);
    v16bf a1 = load_a(sA[p], wm * 32 + 16, 0, 40);
    v16bf b  = load_b_wt(sB[p], wn * 64, 0, 40);
#pragma unroll
    for (int j = 0; j < 4; ++j) {
      v16bf bn = b;
      if (j < 3) bn = load_b_wt(sB[p], wn * 64 + (j + 1) * 16, 0, 40);
      acc[0][j] = wmma_bf16(a0, b, acc[0][j]);
      acc[1][j] = wmma_bf16(a1, b, acc[1][j]);
      b = bn;
    }
    p ^= 1;
  }

  const int nIdx = lane & 15;
  const int rAdd = (lane >> 4) << 3;
#pragma unroll
  for (int i = 0; i < 2; ++i)
#pragma unroll
    for (int j = 0; j < 4; ++j) {
      int col = colBase + wn * 64 + j * 16 + nIdx;
      float bv = bias ? bias[col] : 0.f;
#pragma unroll
      for (int r = 0; r < 8; ++r) {
        int row = rowBase + wm * 32 + i * 16 + rAdd + r;
        float v = acc[i][j][r] + bv;
        if (ACT == 1) v = fmaxf(v, 0.f);
        if (ACT == 2) v = 0.5f * v * (1.f + erff(v * 0.70710678118654752f));
        size_t o = (size_t)row * Nn + col;
        if (WF32)  Cf[o] = v;
        if (WBF16) Cb[o] = f2bf(v);
      }
    }
}

// ------------------- fc_delta stage 1: Linear(3,256)+ReLU -------------------

__global__ __launch_bounds__(256)
void k_stage1(const float* __restrict__ pos, const float* __restrict__ w1,
              const float* __restrict__ b1, __bf16* __restrict__ Hout)
{
  size_t g = (size_t)blockIdx.x * 256 + threadIdx.x;   // (b, n, c)
  int c = (int)(g & 255);
  size_t bn = g >> 8;
  int n = (int)(bn & 255);
  int b = (int)(bn >> 8);
  const float* p = pos + (size_t)b * 3 * N_ + n;
  float a = p[0] * w1[c * 3] + p[N_] * w1[c * 3 + 1] + p[2 * N_] * w1[c * 3 + 2] + b1[c];
  Hout[g] = f2bf(fmaxf(a, 0.f));
}

// --------------- per-batch column sums for closed-form sqd_feat -------------

__global__ __launch_bounds__(256)
void k_colsum(const float* __restrict__ E, float* __restrict__ S, float* __restrict__ Q)
{
  int g = blockIdx.x * 256 + threadIdx.x;              // over B*C
  int f = g & 255, b = g >> 8;
  const float* e = E + (size_t)b * N_ * C_ + f;
  float s = 0.f, q = 0.f;
  for (int n = 0; n < N_; ++n) { float v = e[(size_t)n * C_]; s += v; q += v * v; }
  S[g] = s; Q[g] = q;
}

// -------- build q/k/v (token-major (L,B,E)) + f32 residual copy of q --------

__global__ __launch_bounds__(256)
void k_build_qkv(const float* __restrict__ pcE, const float* __restrict__ kpE,
                 const float* __restrict__ Spc, const float* __restrict__ Qpc,
                 const float* __restrict__ Skp, const float* __restrict__ Qkp,
                 const float* __restrict__ q_in, const float* __restrict__ k_in,
                 const float* __restrict__ v_in,
                 __bf16* __restrict__ Qt, __bf16* __restrict__ Kt,
                 __bf16* __restrict__ Vt, float* __restrict__ qres)
{
  size_t g = (size_t)blockIdx.x * 256 + threadIdx.x;   // (b, l, e)
  int e = (int)(g & 255);
  size_t bl = g >> 8;
  int l = (int)(bl & 255);
  int b = (int)(bl >> 8);
  float pe = pcE[g], ke = kpE[g];
  int bf = b * 256 + e;
  float dpc = 256.f * pe * pe - 2.f * pe * Skp[bf] + Qkp[bf];
  float dkp = 256.f * ke * ke - 2.f * ke * Spc[bf] + Qpc[bf];
  float qv = q_in[g] + dpc;
  float kv = k_in[g] + dkp;
  float vv = v_in[g] + dkp;
  size_t t = ((size_t)l * B_ + b) * C_ + e;
  Qt[t] = f2bf(qv); Kt[t] = f2bf(kv); Vt[t] = f2bf(vv);
  qres[t] = qv;
}

// ----------------- fused attention per (b,h): QK^T/softmax/AV ---------------

// Stage a 256x64 bf16 head tile from token-major global into LDS via DMA.
static __device__ __forceinline__ void stage_head(const __bf16* __restrict__ G,
                                                  int b, int h, __bf16* dst, int tid) {
#pragma unroll
  for (int i = tid; i < 2048; i += 256) {      // 256 rows x 4 chunks(16B) x 2
    int l = i >> 3, sg = i & 7;
    async_b128(G + ((size_t)l * B_ + b) * C_ + h * DH_ + sg * 8,
               dst + l * 64 + sg * 8);
  }
}

__global__ __launch_bounds__(256)
void k_attn(const __bf16* __restrict__ Qp, const __bf16* __restrict__ Kp,
            const __bf16* __restrict__ Vp, __bf16* __restrict__ Ot)
{
  extern __shared__ __bf16 smem[];
  __bf16* sQ   = smem;               // 256 x 64 bf16   (32 KB)
  __bf16* sKV  = smem + 256 * 64;    // 256 x 64 bf16   (32 KB), K then V
  __bf16* sAtt = smem + 2 * 256 * 64;// 256 x 256 bf16  (128 KB)
  const int b = blockIdx.x >> 2;
  const int h = blockIdx.x & 3;
  const int tid = threadIdx.x;
  const int wv = tid >> 5, lane = tid & 31;

  stage_head(Qp, b, h, sQ, tid);
  stage_head(Kp, b, h, sKV, tid);
  wait_async0();
  __syncthreads();

  const float scale = 0.125f;        // 1/sqrt(64)
#pragma unroll 1
  for (int c = 0; c < 2; ++c) {      // each wave: 2 chunks of 16 rows
    int row0 = wv * 32 + c * 16;
    v16bf a0 = load_a(sQ, row0, 0, 64);
    v16bf a1 = load_a(sQ, row0, 32, 64);
    v8f acc[16];
    v16bf b0 = load_b_wt(sKV, 0, 0, 64);
    v16bf b1 = load_b_wt(sKV, 0, 32, 64);
#pragma unroll
    for (int mt = 0; mt < 16; ++mt) {          // pipelined B-fragment loads
      v16bf c0 = b0, c1 = b1;
      if (mt < 15) {
        c0 = load_b_wt(sKV, (mt + 1) * 16, 0, 64);
        c1 = load_b_wt(sKV, (mt + 1) * 16, 32, 64);
      }
      acc[mt] = v8f{};
      acc[mt] = wmma_bf16(a0, b0, acc[mt]);
      acc[mt] = wmma_bf16(a1, b1, acc[mt]);
      b0 = c0; b1 = c1;
    }
    float mx[8], sm[8];
#pragma unroll
    for (int r = 0; r < 8; ++r) mx[r] = -1e30f;
#pragma unroll
    for (int mt = 0; mt < 16; ++mt)
#pragma unroll
      for (int r = 0; r < 8; ++r) mx[r] = fmaxf(mx[r], acc[mt][r] * scale);
#pragma unroll
    for (int r = 0; r < 8; ++r) { mx[r] = half16_max(mx[r]); sm[r] = 0.f; }
#pragma unroll
    for (int mt = 0; mt < 16; ++mt)
#pragma unroll
      for (int r = 0; r < 8; ++r) {
        float p = __expf(acc[mt][r] * scale - mx[r]);
        acc[mt][r] = p; sm[r] += p;
      }
#pragma unroll
    for (int r = 0; r < 8; ++r) sm[r] = 1.f / half16_sum(sm[r]);
    const int nIdx = lane & 15, rAdd = (lane >> 4) << 3;
#pragma unroll
    for (int mt = 0; mt < 16; ++mt)
#pragma unroll
      for (int r = 0; r < 8; ++r)
        sAtt[(row0 + rAdd + r) * 256 + mt * 16 + nIdx] = f2bf(acc[mt][r] * sm[r]);
  }
  __syncthreads();

  stage_head(Vp, b, h, sKV, tid);    // overwrite K tile with V
  wait_async0();
  __syncthreads();

#pragma unroll 1
  for (int c = 0; c < 2; ++c) {      // O = att @ V
    int row0 = wv * 32 + c * 16;
    v8f acc[4];
#pragma unroll
    for (int j = 0; j < 4; ++j) acc[j] = v8f{};
#pragma unroll 1
    for (int kt = 0; kt < 8; ++kt) {
      v16bf a = load_a(sAtt, row0, kt * 32, 256);
      v16bf bb = load_b_rm(sKV, kt * 32, 0, 64);
#pragma unroll
      for (int j = 0; j < 4; ++j) {
        v16bf bn = bb;
        if (j < 3) bn = load_b_rm(sKV, kt * 32, (j + 1) * 16, 64);
        acc[j] = wmma_bf16(a, bb, acc[j]);
        bb = bn;
      }
    }
    const int nIdx = lane & 15, rAdd = (lane >> 4) << 3;
#pragma unroll
    for (int j = 0; j < 4; ++j)
#pragma unroll
      for (int r = 0; r < 8; ++r) {
        int l = row0 + rAdd + r;
        Ot[((size_t)l * B_ + b) * C_ + h * DH_ + j * 16 + nIdx] = f2bf(acc[j][r]);
      }
  }
}

// ---------------------- residual + LayerNorm(256) ---------------------------

__global__ __launch_bounds__(256)
void k_add_ln(const float* __restrict__ qres, const float* __restrict__ xO,
              const float* __restrict__ g, const float* __restrict__ bta,
              float* __restrict__ xln, __bf16* __restrict__ xlnb)
{
  __shared__ float red[256];
  const int t = blockIdx.x, e = threadIdx.x;
  size_t o = (size_t)t * 256 + e;
  float x = qres[o] + xO[o];
  red[e] = x; __syncthreads();
  for (int s = 128; s; s >>= 1) { if (e < s) red[e] += red[e + s]; __syncthreads(); }
  float mu = red[0] * (1.f / 256.f);
  __syncthreads();
  float d = x - mu;
  red[e] = d * d; __syncthreads();
  for (int s = 128; s; s >>= 1) { if (e < s) red[e] += red[e + s]; __syncthreads(); }
  float var = red[0] * (1.f / 256.f);
  float y = d * rsqrtf(var + 1e-5f) * g[e] + bta[e];
  xln[o] = y; xlnb[o] = f2bf(y);
}

// ------------------- final residual + transpose to (B,C,L) ------------------

__global__ __launch_bounds__(256)
void k_final(const float* __restrict__ xln, const float* __restrict__ ffO,
             float* __restrict__ out)
{
  size_t g = (size_t)blockIdx.x * 256 + threadIdx.x;   // (b, e, l)
  int l = (int)(g & 255);
  size_t be = g >> 8;
  int e = (int)(be & 255);
  int b = (int)(be >> 8);
  size_t src = ((size_t)l * B_ + b) * C_ + e;
  out[g] = xln[src] + ffO[src];
}

// ----------------------------- f32 -> bf16 ----------------------------------

__global__ __launch_bounds__(256)
void k_f2bf(const float* __restrict__ in, __bf16* __restrict__ out, int n)
{
  int i = blockIdx.x * 256 + threadIdx.x;
  if (i < n) out[i] = f2bf(in[i]);
}

// ------------------------------- launcher -----------------------------------

extern "C" void kernel_launch(void* const* d_in, const int* in_sizes, int n_in,
                              void* d_out, int out_size, void* d_ws, size_t ws_size,
                              hipStream_t stream) {
  const float* pos_pc  = (const float*)d_in[0];
  const float* pos_kp  = (const float*)d_in[1];
  const float* key_in  = (const float*)d_in[2];
  const float* query_in= (const float*)d_in[3];
  const float* value_in= (const float*)d_in[4];
  // d_in[5] = upfeat (dead arg)
  const float* fcd_w1  = (const float*)d_in[6];
  const float* fcd_b1  = (const float*)d_in[7];
  const float* fcd_w2  = (const float*)d_in[8];
  const float* fcd_b2  = (const float*)d_in[9];
  const float* wq = (const float*)d_in[10]; const float* bq = (const float*)d_in[11];
  const float* wk = (const float*)d_in[12]; const float* bk = (const float*)d_in[13];
  const float* wv = (const float*)d_in[14]; const float* bv = (const float*)d_in[15];
  const float* wo = (const float*)d_in[16]; const float* bo = (const float*)d_in[17];
  const float* ln_g = (const float*)d_in[18]; const float* ln_b = (const float*)d_in[19];
  const float* w11 = (const float*)d_in[20]; const float* b11 = (const float*)d_in[21];
  const float* w12 = (const float*)d_in[22]; const float* b12 = (const float*)d_in[23];
  float* out = (float*)d_out;

  char* ws = (char*)d_ws;
  const size_t MB32 = (size_t)T_ * C_ * sizeof(__bf16);   // 33,554,432
  const size_t MB64 = (size_t)T_ * C_ * sizeof(float);    // 67,108,864
  // Region R1 (134 MB): stage-1 activations, later FFN hidden h1
  char* R1 = ws;
  __bf16* hPC = (__bf16*)R1;
  __bf16* hKP = (__bf16*)(R1 + MB32);
  __bf16* h1  = (__bf16*)R1;                               // T_ x FF_ bf16
  // Region R2 (134 MB): pcE/kpE, later Qp/Kp/Vp, later xln/xlnb
  char* R2 = R1 + 2 * MB32 * 2;   // 134,217,728
  float* pcE = (float*)R2;
  float* kpE = (float*)(R2 + MB64);
  __bf16* Qp = (__bf16*)R2;
  __bf16* Kp = (__bf16*)(R2 + MB32);
  __bf16* Vp = (__bf16*)(R2 + 2 * MB32);
  float*  xln  = (float*)R2;
  __bf16* xlnb = (__bf16*)(R2 + MB64);
  // Region R3 (100.7 MB): Qtok/Ktok/Vtok, later Otok + xO
  char* R3 = R2 + 2 * MB64;
  __bf16* Qt = (__bf16*)R3;
  __bf16* Kt = (__bf16*)(R3 + MB32);
  __bf16* Vt = (__bf16*)(R3 + 2 * MB32);
  __bf16* Ot = (__bf16*)R3;
  float*  xO = (float*)(R3 + MB32);
  // Region R4 (67 MB): qres, later ffO
  char* R4 = R3 + 3 * MB32;
  float* qres = (float*)R4;
  float* ffO  = (float*)R4;
  // Region R5: per-batch sums
  char* R5 = R4 + MB64;
  float* Spc = (float*)R5;
  float* Qpc = Spc + B_ * C_;
  float* Skp = Qpc + B_ * C_;
  float* Qkp = Skp + B_ * C_;
  // Region R6: bf16 weights
  char* R6 = R5 + 4 * B_ * C_ * sizeof(float);
  __bf16* w2b = (__bf16*)R6;
  __bf16* wqb = w2b + C_ * C_;
  __bf16* wkb = wqb + C_ * C_;
  __bf16* wvb = wkb + C_ * C_;
  __bf16* wob = wvb + C_ * C_;
  __bf16* w11b = wob + C_ * C_;
  __bf16* w12b = w11b + (size_t)FF_ * C_;
  (void)ws_size; (void)n_in; (void)in_sizes; (void)out_size;

  // 0) weight conversions to bf16
  k_f2bf<<<C_ * C_ / 256, 256, 0, stream>>>(fcd_w2, w2b, C_ * C_);
  k_f2bf<<<C_ * C_ / 256, 256, 0, stream>>>(wq, wqb, C_ * C_);
  k_f2bf<<<C_ * C_ / 256, 256, 0, stream>>>(wk, wkb, C_ * C_);
  k_f2bf<<<C_ * C_ / 256, 256, 0, stream>>>(wv, wvb, C_ * C_);
  k_f2bf<<<C_ * C_ / 256, 256, 0, stream>>>(wo, wob, C_ * C_);
  k_f2bf<<<FF_ * C_ / 256, 256, 0, stream>>>(w11, w11b, FF_ * C_);
  k_f2bf<<<FF_ * C_ / 256, 256, 0, stream>>>(w12, w12b, FF_ * C_);

  // 1) fc_delta stage 1 (K=3 VALU) for both point clouds
  k_stage1<<<T_ * C_ / 256, 256, 0, stream>>>(pos_pc, fcd_w1, fcd_b1, hPC);
  k_stage1<<<T_ * C_ / 256, 256, 0, stream>>>(pos_kp, fcd_w1, fcd_b1, hKP);

  // 2) fc_delta stage 2: (65536 x 256) @ (256 x 256)^T  -> pcE / kpE (f32)
  dim3 g256(C_ / 128, T_ / 128);
  k_gemm<0, true, false><<<g256, 256, 0, stream>>>(hPC, w2b, fcd_b2, pcE, nullptr, T_, C_, C_);
  k_gemm<0, true, false><<<g256, 256, 0, stream>>>(hKP, w2b, fcd_b2, kpE, nullptr, T_, C_, C_);

  // 3) closed-form sqd_feat reductions
  k_colsum<<<B_ * C_ / 256, 256, 0, stream>>>(pcE, Spc, Qpc);
  k_colsum<<<B_ * C_ / 256, 256, 0, stream>>>(kpE, Skp, Qkp);

  // 4) build q,k,v tokens (L,B,E layout) + residual
  k_build_qkv<<<T_ * C_ / 256, 256, 0, stream>>>(pcE, kpE, Spc, Qpc, Skp, Qkp,
                                                 query_in, key_in, value_in,
                                                 Qt, Kt, Vt, qres);

  // 5) in-projections (bf16 out for attention)
  k_gemm<0, false, true><<<g256, 256, 0, stream>>>(Qt, wqb, bq, nullptr, Qp, T_, C_, C_);
  k_gemm<0, false, true><<<g256, 256, 0, stream>>>(Kt, wkb, bk, nullptr, Kp, T_, C_, C_);
  k_gemm<0, false, true><<<g256, 256, 0, stream>>>(Vt, wvb, bv, nullptr, Vp, T_, C_, C_);

  // 6) fused attention, one block per (b,h); 192 KB dynamic LDS (<= 320 KB WGP)
  const size_t attn_lds = (size_t)(2 * 256 * 64 + 256 * 256) * sizeof(__bf16);
  hipFuncSetAttribute((const void*)k_attn,
                      hipFuncAttributeMaxDynamicSharedMemorySize, (int)attn_lds);
  k_attn<<<B_ * H_, 256, attn_lds, stream>>>(Qp, Kp, Vp, Ot);

  // 7) out-projection (f32)
  k_gemm<0, true, false><<<g256, 256, 0, stream>>>(Ot, wob, bo, xO, nullptr, T_, C_, C_);

  // 8) residual + LayerNorm
  k_add_ln<<<T_, 256, 0, stream>>>(qres, xO, ln_g, ln_b, xln, xlnb);

  // 9) FFN: GELU(x @ w11^T + b11) @ w12^T + b12
  dim3 gff1(FF_ / 128, T_ / 128);
  k_gemm<2, false, true><<<gff1, 256, 0, stream>>>(xlnb, w11b, b11, nullptr, h1, T_, FF_, C_);
  k_gemm<0, true, false><<<g256, 256, 0, stream>>>(h1, w12b, b12, ffO, nullptr, T_, C_, FF_);

  // 10) final residual + transpose to (B, C, L)
  k_final<<<T_ * C_ / 256, 256, 0, stream>>>(xln, ffO, out);
}
